// MultiHeadSelfAttention_49795850829899
// MI455X (gfx1250) — matrix operand order repaired and checked by reading, not compiled, not run on hardware
//
#include <hip/hip_runtime.h>
#include <hip/hip_bf16.h>

// ---------------- problem constants ----------------
constexpr int NB  = 4;
constexpr int NS  = 2048;
constexpr int ND  = 1024;
constexpr int NH  = 16;
constexpr int NDH = 64;       // head dim
constexpr int MT  = NB * NS;  // 8192 rows

typedef __attribute__((ext_vector_type(16))) __bf16          v16bf;
typedef __attribute__((ext_vector_type(8)))  float           v8f;
typedef __attribute__((ext_vector_type(16))) unsigned short  v16u;
typedef __attribute__((ext_vector_type(8)))  unsigned short  v8u;
typedef __attribute__((ext_vector_type(4)))  unsigned int    v4ui;
typedef __attribute__((ext_vector_type(8)))  unsigned int    v8ui;

// float -> bf16 bits, round-to-nearest-even
__device__ __forceinline__ unsigned short f2bf(float f) {
  unsigned int u = __builtin_bit_cast(unsigned int, f);
  unsigned int r = u + 0x7FFFu + ((u >> 16) & 1u);
  return (unsigned short)(r >> 16);
}

__device__ __forceinline__ v8f wmma_bf16(v16u a, v16u b, v8f c) {
  return __builtin_amdgcn_wmma_f32_16x16x32_bf16(
      false, __builtin_bit_cast(v16bf, a),
      false, __builtin_bit_cast(v16bf, b),
      (short)0, c, false, false);
}

// gather two aligned 8x-bf16 runs into one 16-element fragment register
__device__ __forceinline__ v16u ld16(const unsigned short* p0,
                                     const unsigned short* p1) {
  v8u a = *(const v8u*)p0;
  v8u b = *(const v8u*)p1;
  v16u r;
#pragma unroll
  for (int i = 0; i < 8; i++) { r[i] = a[i]; r[i + 8] = b[i]; }
  return r;
}

// ---------------------------------------------------------------------
// TDM: DMA a 2D bf16 tile (tile_d0 x tile_d1) from global into LDS with
// per-row padding (D# per CDNA5 ISA: group0 = count/lds/global/type,
// group1 = data_size|pad, dims, tile dims, stride). Issued via inline asm
// (portable across toolchains); completion via s_wait_tensorcnt.
// ---------------------------------------------------------------------
__device__ __forceinline__ void tdm_load_2d_bf16(
    unsigned lds_addr, const void* gaddr,
    unsigned tensor_d0, unsigned tensor_d1,
    unsigned tile_d0, unsigned tile_d1,
    unsigned stride0, unsigned pad_interval_code, unsigned pad_amount_code) {
  unsigned long long ga = (unsigned long long)gaddr;
  v4ui g0;
  g0[0] = 1u;                                   // count=1, user descriptor
  g0[1] = lds_addr;                             // LDS byte address
  g0[2] = (unsigned)ga;                         // global_addr[31:0]
  g0[3] = (unsigned)(ga >> 32) | (2u << 30);    // global_addr[56:32] | type=2
  v8ui g1;
  g1[0] = (1u << 16)                            // data_size = 2 bytes
        | (1u << 20)                            // pad_enable
        | (pad_interval_code << 22)             // DWORDs per row before pad
        | (pad_amount_code << 25);              // pad DWORDs
  g1[1] = (tensor_d0 & 0xffffu) << 16;                                  // dim0 lo
  g1[2] = ((tensor_d0 >> 16) & 0xffffu) | ((tensor_d1 & 0xffffu) << 16);
  g1[3] = ((tensor_d1 >> 16) & 0xffffu) | ((tile_d0 & 0xffffu) << 16);
  g1[4] = tile_d1 & 0xffffu;                    // tile_dim1 (tile_dim2 = 0)
  g1[5] = stride0;                              // tensor_dim0_stride[31:0]
  g1[6] = 0u;
  g1[7] = 0u;
  v4ui gz = {0u, 0u, 0u, 0u};                   // groups 2/3 unused (2D)
  asm volatile("tensor_load_to_lds %0, %1, %2, %3"
               :: "s"(g0), "s"(g1), "s"(gz), "s"(gz)
               : "memory");
}

// =====================================================================
// Kernel 1: fused QKV projection.  out = x @ W + bias, stored bf16 in
// head-major [B,H,S,DH].  grid = (MT/128, ND/64, 3); block = 256.
// (A needs f32->bf16 conversion, so staging stays VALU-based here.)
// =====================================================================
__global__ __launch_bounds__(256) void k_qkv(
    const float* __restrict__ x,
    const float* __restrict__ wq, const float* __restrict__ biasq,
    const float* __restrict__ wk, const float* __restrict__ biask,
    const float* __restrict__ wv, const float* __restrict__ biasv,
    unsigned short* __restrict__ Qh,
    unsigned short* __restrict__ Kh,
    unsigned short* __restrict__ Vh) {
  __shared__ __align__(16) unsigned short As[128][40];  // M x K (bf16)
  __shared__ __align__(16) unsigned short Bs[64][40];   // N x K (transposed)

  const int t = threadIdx.x;
  const int z = blockIdx.z;
  const float* w    = (z == 0) ? wq : (z == 1) ? wk : wv;
  const float* bias = (z == 0) ? biasq : (z == 1) ? biask : biasv;
  unsigned short* out = (z == 0) ? Qh : (z == 1) ? Kh : Vh;

  const int m0 = blockIdx.x * 128;
  const int n0 = blockIdx.y * 64;

  const int wave = t >> 5, lane = t & 31;
  const int wm = (wave & 3) * 32;
  const int wn = (wave >> 2) * 32;
  const int lr = lane & 15;
  const int lh = lane >> 4;
  const int abase = lh * 8;

  v8f acc[2][2] = {};

  for (int kt = 0; kt < ND; kt += 32) {
    __syncthreads();
#pragma unroll
    for (int i = 0; i < 16; i++) {
      int e = t + i * 256;
      int r = e >> 5, c = e & 31;
      As[r][c] = f2bf(x[(size_t)(m0 + r) * ND + kt + c]);
    }
#pragma unroll
    for (int i = 0; i < 8; i++) {
      int e = t + i * 256;
      int k = e >> 6, n = e & 63;
      Bs[n][k] = f2bf(w[(size_t)(kt + k) * ND + n0 + n]);
    }
    __syncthreads();

    v16u afr[2], bfr[2];
#pragma unroll
    for (int ai = 0; ai < 2; ai++) {
      int row = wm + ai * 16 + lr;
      afr[ai] = ld16(&As[row][abase], &As[row][abase + 16]);
    }
#pragma unroll
    for (int bi = 0; bi < 2; bi++) {
      int col = wn + bi * 16 + lr;
      bfr[bi] = ld16(&Bs[col][lh * 8], &Bs[col][lh * 8 + 16]);
    }
#pragma unroll
    for (int ai = 0; ai < 2; ai++)
#pragma unroll
      for (int bi = 0; bi < 2; bi++)
        acc[ai][bi] = wmma_bf16(afr[ai], bfr[bi], acc[ai][bi]);
  }

#pragma unroll
  for (int ai = 0; ai < 2; ai++) {
    int mbase = m0 + wm + ai * 16 + lh * 8;
#pragma unroll
    for (int bi = 0; bi < 2; bi++) {
      int n = n0 + wn + bi * 16 + lr;
      float bv = bias[n];
      int h = n >> 6, dh = n & (NDH - 1);
#pragma unroll
      for (int r = 0; r < 8; r++) {
        int mg = mbase + r;
        int b = mg / NS, s = mg & (NS - 1);
        size_t off = (((size_t)b * NH + h) * NS + s) * NDH + dh;
        out[off] = f2bf(acc[ai][bi][r] + bv);
      }
    }
  }
}

// =====================================================================
// Kernel 2: flash attention. grid = (NS/128, NH, NB); block = 256.
// K tile staged by the Tensor Data Mover (row of 64 bf16 = 32 DWORDs,
// pad 4 DWORDs -> LDS stride 72); V tile cooperatively transposed.
// =====================================================================
__global__ __launch_bounds__(256) void k_attn(
    const unsigned short* __restrict__ Qh,
    const unsigned short* __restrict__ Kh,
    const unsigned short* __restrict__ Vh,
    const unsigned char* __restrict__ mask,
    unsigned short* __restrict__ Ctx) {
  __shared__ __align__(16) unsigned short Ks[32][72];      // [krow][d]
  __shared__ __align__(16) unsigned short Vt[64][40];      // [d][krow]
  __shared__ __align__(16) unsigned short Ps[8][16][40];   // per-wave P tile

  const int t = threadIdx.x;
  const int wave = t >> 5, lane = t & 31;
  const int lr = lane & 15, lh = lane >> 4;
  const int b = blockIdx.z, h = blockIdx.y;
  const int q0 = blockIdx.x * 128 + wave * 16;

  const size_t bh = ((size_t)b * NH + h) * NS;  // row base in [B,H,S,DH]
  const unsigned ldsK = (unsigned)(uintptr_t)&Ks[0][0];

  // Q fragments: 2 chunks over DH (K=32 each)
  v16u qf[2];
  {
    const unsigned short* qp = Qh + (bh + q0 + lr) * NDH;
#pragma unroll
    for (int c = 0; c < 2; c++)
      qf[c] = ld16(qp + c * 32 + lh * 8, qp + c * 32 + lh * 8 + 16);
  }

  float mrow[8], lsum[8];
  v8f acc[4] = {};
#pragma unroll
  for (int r = 0; r < 8; r++) { mrow[r] = -1e30f; lsum[r] = 0.0f; }

  const float scale = 0.03125f;  // 1/sqrt(D) = 1/32

  for (int j = 0; j < NS; j += 32) {
    __syncthreads();
    // async TDM load of K tile rows j..j+31 into padded LDS
    if (wave == 0) {
      tdm_load_2d_bf16(ldsK, Kh + (bh + j) * NDH,
                       /*tensor_d0=*/NDH, /*tensor_d1=*/(unsigned)(NS - j),
                       /*tile_d0=*/NDH, /*tile_d1=*/32,
                       /*stride0=*/NDH,
                       /*pad_interval=*/4u /*32 DW*/, /*pad_amount=*/3u /*4 DW*/);
    }
    // meanwhile: cooperative V tile transpose into LDS
#pragma unroll
    for (int i = 0; i < 4; i++) {
      int e = t + i * 256;           // dword index, 1024 total
      int kr = e >> 5;
      int d2 = (e & 31) * 2;
      unsigned int vv = *(const unsigned int*)(Vh + (bh + j + kr) * NDH + d2);
      Vt[d2][kr]     = (unsigned short)(vv & 0xffffu);
      Vt[d2 + 1][kr] = (unsigned short)(vv >> 16);
    }
    if (j + 32 < NS) {  // prefetch next tiles into cache
      __builtin_prefetch(Kh + (bh + j + 32) * NDH, 0, 0);
      __builtin_prefetch(Vh + (bh + j + 32) * NDH, 0, 0);
    }
    if (wave == 0) asm volatile("s_wait_tensorcnt 0x0" ::: "memory");
    __syncthreads();

    // scores: two 16x16 tiles covering keys j..j+31
    v8f st[2];
#pragma unroll
    for (int tt = 0; tt < 2; tt++) {
      v8f s = {};
#pragma unroll
      for (int c = 0; c < 2; c++) {
        v16u bf = ld16(&Ks[tt * 16 + lr][c * 32 + lh * 8],
                       &Ks[tt * 16 + lr][c * 32 + lh * 8 + 16]);
        s = wmma_bf16(qf[c], bf, s);
      }
      st[tt] = s;
    }

    // scale + key mask (column depends only on lane, not on row)
    float mk[2];
#pragma unroll
    for (int tt = 0; tt < 2; tt++) {
      int col = j + tt * 16 + lr;
      mk[tt] = mask[(size_t)b * NS + col] ? 0.0f : -1e9f;
    }
#pragma unroll
    for (int tt = 0; tt < 2; tt++)
#pragma unroll
      for (int r = 0; r < 8; r++) st[tt][r] = st[tt][r] * scale + mk[tt];

    // online softmax (rows stripe across 16-lane groups)
    float alpha[8];
#pragma unroll
    for (int r = 0; r < 8; r++) {
      float v = fmaxf(st[0][r], st[1][r]);
#pragma unroll
      for (int off = 1; off < 16; off <<= 1)
        v = fmaxf(v, __shfl_xor(v, off));
      float mn = fmaxf(mrow[r], v);
      alpha[r] = __expf(mrow[r] - mn);
      mrow[r] = mn;
      float p0 = __expf(st[0][r] - mn);
      float p1 = __expf(st[1][r] - mn);
      st[0][r] = p0; st[1][r] = p1;
      float rs = p0 + p1;
#pragma unroll
      for (int off = 1; off < 16; off <<= 1) rs += __shfl_xor(rs, off);
      lsum[r] = lsum[r] * alpha[r] + rs;
    }

    // D-layout -> A-layout transpose of P through wave-private LDS
#pragma unroll
    for (int tt = 0; tt < 2; tt++)
#pragma unroll
      for (int r = 0; r < 8; r++)
        Ps[wave][lh * 8 + r][tt * 16 + lr] = f2bf(st[tt][r]);
    asm volatile("s_wait_dscnt 0" ::: "memory");  // DS in-order within wave

    v16u pf = ld16(&Ps[wave][lr][lh * 8], &Ps[wave][lr][lh * 8 + 16]);

    // P @ V with online rescale of accumulators
#pragma unroll
    for (int f = 0; f < 4; f++) {
      v16u vf = ld16(&Vt[f * 16 + lr][lh * 8], &Vt[f * 16 + lr][lh * 8 + 16]);
#pragma unroll
      for (int r = 0; r < 8; r++) acc[f][r] *= alpha[r];
      acc[f] = wmma_bf16(pf, vf, acc[f]);
    }
  }

  // finalize: ctx[b][s][h*DH+dh] bf16 (merged-head, row-major D)
#pragma unroll
  for (int f = 0; f < 4; f++)
#pragma unroll
    for (int r = 0; r < 8; r++) {
      int row = q0 + lh * 8 + r;
      int dh = f * 16 + lr;
      float o = acc[f][r] / lsum[r];
      Ctx[((size_t)b * NS + row) * ND + h * NDH + dh] = f2bf(o);
    }
}

// =====================================================================
// Kernel 3: output projection. out = ctx @ Wo + bo (f32 out).
// A tile (bf16 row-major copy) staged by TDM; B converted in VALU.
// grid = (MT/128, ND/64); block = 256.
// =====================================================================
__global__ __launch_bounds__(256) void k_oproj(
    const unsigned short* __restrict__ Ctx,
    const float* __restrict__ wo, const float* __restrict__ bo,
    float* __restrict__ out) {
  __shared__ __align__(16) unsigned short As[128][40];
  __shared__ __align__(16) unsigned short Bs[64][40];

  const int t = threadIdx.x;
  const int m0 = blockIdx.x * 128;
  const int n0 = blockIdx.y * 64;
  const int wave = t >> 5, lane = t & 31;
  const int wm = (wave & 3) * 32, wn = (wave >> 2) * 32;
  const int lr = lane & 15, lh = lane >> 4;
  const int abase = lh * 8;
  const unsigned ldsA = (unsigned)(uintptr_t)&As[0][0];

  v8f acc[2][2] = {};

  for (int kt = 0; kt < ND; kt += 32) {
    __syncthreads();
    // A tile via TDM: 32 cols x 128 rows, row = 16 DWORDs + 4 DWORD pad -> stride 40
    if (wave == 0) {
      tdm_load_2d_bf16(ldsA, Ctx + (size_t)m0 * ND + kt,
                       /*tensor_d0=*/(unsigned)(ND - kt), /*tensor_d1=*/(unsigned)(MT - m0),
                       /*tile_d0=*/32, /*tile_d1=*/128,
                       /*stride0=*/ND,
                       /*pad_interval=*/3u /*16 DW*/, /*pad_amount=*/3u /*4 DW*/);
    }
#pragma unroll
    for (int i = 0; i < 8; i++) {        // B: f32 -> bf16 transposed
      int e = t + i * 256;
      int k = e >> 6, n = e & 63;
      Bs[n][k] = f2bf(wo[(size_t)(kt + k) * ND + n0 + n]);
    }
    if (wave == 0) asm volatile("s_wait_tensorcnt 0x0" ::: "memory");
    __syncthreads();

    v16u afr[2], bfr[2];
#pragma unroll
    for (int ai = 0; ai < 2; ai++) {
      int row = wm + ai * 16 + lr;
      afr[ai] = ld16(&As[row][abase], &As[row][abase + 16]);
    }
#pragma unroll
    for (int bi = 0; bi < 2; bi++) {
      int col = wn + bi * 16 + lr;
      bfr[bi] = ld16(&Bs[col][lh * 8], &Bs[col][lh * 8 + 16]);
    }
#pragma unroll
    for (int ai = 0; ai < 2; ai++)
#pragma unroll
      for (int bi = 0; bi < 2; bi++)
        acc[ai][bi] = wmma_bf16(afr[ai], bfr[bi], acc[ai][bi]);
  }

#pragma unroll
  for (int ai = 0; ai < 2; ai++) {
    int mbase = m0 + wm + ai * 16 + lh * 8;
#pragma unroll
    for (int bi = 0; bi < 2; bi++) {
      int n = n0 + wn + bi * 16 + lr;
      float bv = bo[n];
#pragma unroll
      for (int r = 0; r < 8; r++)
        out[(size_t)(mbase + r) * ND + n] = acc[ai][bi][r] + bv;
    }
  }
}

// =====================================================================
extern "C" void kernel_launch(void* const* d_in, const int* in_sizes, int n_in,
                              void* d_out, int out_size, void* d_ws,
                              size_t ws_size, hipStream_t stream) {
  (void)in_sizes; (void)n_in; (void)out_size; (void)ws_size;
  const float* x          = (const float*)d_in[0];
  const unsigned char* mk = (const unsigned char*)d_in[1];
  const float* wq = (const float*)d_in[2];
  const float* bq = (const float*)d_in[3];
  const float* wk = (const float*)d_in[4];
  const float* bk = (const float*)d_in[5];
  const float* wv = (const float*)d_in[6];
  const float* bv = (const float*)d_in[7];
  const float* wo = (const float*)d_in[8];
  const float* bo = (const float*)d_in[9];
  float* out = (float*)d_out;

  char* ws = (char*)d_ws;
  const size_t seg = (size_t)NB * NH * NS * NDH * sizeof(unsigned short); // 16 MB
  unsigned short* Qh  = (unsigned short*)(ws);
  unsigned short* Kh  = (unsigned short*)(ws + seg);
  unsigned short* Vh  = (unsigned short*)(ws + 2 * seg);
  unsigned short* Ctx = (unsigned short*)(ws + 3 * seg);

  dim3 blk(256);
  k_qkv <<<dim3(MT / 128, ND / 64, 3), blk, 0, stream>>>(
      x, wq, bq, wk, bk, wv, bv, Qh, Kh, Vh);
  k_attn<<<dim3(NS / 128, NH, NB), blk, 0, stream>>>(Qh, Kh, Vh, mk, Ctx);
  k_oproj<<<dim3(MT / 128, ND / 64), blk, 0, stream>>>(Ctx, wo, bo, out);
}
